// Encoder_Motion_Attn_Res_35141422415984
// MI455X (gfx1250) — matrix operand order repaired and checked by reading, not compiled, not run on hardware
//
#include <hip/hip_runtime.h>
#include <hip/hip_fp16.h>
#include <math.h>

// ---------------------------------------------------------------------------
// MI455X / gfx1250 Encoder_Motion_Attn_Res.
// All matmuls (convs via implicit GEMM, attention projections) run through
// v_wmma_f32_16x16x32_f16. Block tile 128x128, wave tile 32x64 (2x4 WMMA).
// Tiles are staged to LDS by the Tensor Data Mover with double-buffered
// issue-ahead (TENSORcnt pipelining); manual staging as fallback.
// ---------------------------------------------------------------------------

typedef __attribute__((ext_vector_type(16))) _Float16 v16h;
typedef __attribute__((ext_vector_type(8)))  _Float16 v8h;
typedef __attribute__((ext_vector_type(8)))  float    v8f;
typedef __attribute__((ext_vector_type(4)))  unsigned u32x4;
typedef __attribute__((ext_vector_type(8)))  int      i32x8;
typedef __attribute__((ext_vector_type(4)))  int      i32x4;

#define DEVFN static __device__ __forceinline__

#if defined(__AMDGCN__) && __has_builtin(__builtin_amdgcn_tensor_load_to_lds) && \
    __has_builtin(__builtin_amdgcn_s_wait_tensorcnt)
#define HAVE_TDM 1
#else
#define HAVE_TDM 0
#endif

// ---- WMMA fragments (ISA 7.12.2 16-bit layouts), LDS tiles [row][32 k] ----
// A: lane(m,h) -> e<8: k=8h+e ; e>=8: k=16+8h+(e-8)  => two contiguous 16B runs
DEVFN v16h frag_a(const _Float16* __restrict__ As, int row0) {
  const int lane = threadIdx.x & 31;
  const _Float16* r = As + (size_t)(row0 + (lane & 15)) * 32 + 8 * (lane >> 4);
  v8h lo = *(const v8h*)r;          // k = 8h .. 8h+7
  v8h hi = *(const v8h*)(r + 16);   // k = 16+8h .. 16+8h+7
  return __builtin_shufflevector(lo, hi, 0, 1, 2, 3, 4, 5, 6, 7,
                                 8, 9, 10, 11, 12, 13, 14, 15);
}
// B tile stored [n][k] (weights packed transposed): same k order as A (two 16B runs)
DEVFN v16h frag_b(const _Float16* __restrict__ Bs, int col0) {
  const int lane = threadIdx.x & 31;
  const _Float16* r = Bs + (size_t)(col0 + (lane & 15)) * 32 + 8 * (lane >> 4);
  v8h lo = *(const v8h*)r;
  v8h hi = *(const v8h*)(r + 16);
  return __builtin_shufflevector(lo, hi, 0, 1, 2, 3, 4, 5, 6, 7,
                                 8, 9, 10, 11, 12, 13, 14, 15);
}

#if HAVE_TDM
// 2D TDM tile load: tensor [dim1 rows][dim0 cols] f16, tile [tile1 x 32] at ga.
DEVFN void tdm_load_tile(unsigned lds_addr, unsigned long long ga,
                         int dim0, int dim1, int tile1, int stride0) {
  u32x4 g0 = {1u,                                   // count=1 user descriptor
              lds_addr,                             // lds_addr (bytes)
              (unsigned)ga,                         // global_addr[31:0]
              (unsigned)(ga >> 32) | (2u << 30)};   // ga[56:32] | type=2
  i32x8 g1 = {(int)(1u << 16),                      // data_size = 2B
              (int)(((unsigned)dim0 & 0xFFFFu) << 16),
              (int)(((unsigned)dim0 >> 16) | (((unsigned)dim1 & 0xFFFFu) << 16)),
              (int)(((unsigned)dim1 >> 16) | (32u << 16)),   // tile_dim0 = 32
              tile1,                                          // tile_dim1
              stride0,                                        // dim0 stride lo
              0, 0};
  i32x4 z4 = {0, 0, 0, 0};
#if __clang_major__ >= 23
  i32x8 z8 = {0, 0, 0, 0, 0, 0, 0, 0};
  __builtin_amdgcn_tensor_load_to_lds(g0, g1, z4, z4, z8, 0);
#else
  __builtin_amdgcn_tensor_load_to_lds(g0, g1, z4, z4, 0);
#endif
}
#endif

// ---------------------------------------------------------------------------
// Generic WMMA GEMM / implicit-GEMM conv.
//   C[M,N] = A[M,K] * Bt[N,K]^T (+bias[n]) (+relu)
// Bt packed TRANSPOSED in global: Bt[n*K + k].
// IM2COL==0: A row-major [M,K]; C row-major [M,N]; A+B tiles DMA'd by TDM.
// IM2COL==1: A is NCHW f16, k=(kh*KW+kw)*Ci+ci; C NCHW; B tile DMA'd by TDM.
// Spatial dims are powers of two: wout_sh=log2(Wout), hwo_sh=log2(Hout*Wout).
// ---------------------------------------------------------------------------
template <int IM2COL>
__global__ __launch_bounds__(256) void k_gemm_wmma(
    const _Float16* __restrict__ A, const _Float16* __restrict__ Bt,
    const float* __restrict__ bias, float* __restrict__ Cf,
    _Float16* __restrict__ Ch,
    int M, int N, int K,
    int Ci, int Hin, int Win, int wout_sh, int hwo_sh,
    int KW, int stride, int pad, int relu_out)
{
  __shared__ alignas(64) _Float16 As[2][128 * 32];
  __shared__ alignas(64) _Float16 Bs[2][128 * 32];

  const int tid  = threadIdx.x;
  const int wave = tid >> 5;
  const int wm   = wave & 3;   // 4 M-groups of 32 rows
  const int wn   = wave >> 2;  // 2 N-groups of 64 cols
  const int m0   = blockIdx.y * 128;
  const int n0   = blockIdx.x * 128;

  const int kk    = tid & 31;  // staging: thread owns fixed k within tile
  const int rbase = tid >> 5;

  if (bias) __builtin_prefetch(&bias[n0 + (tid & 127)], 0, 2);  // global_prefetch_b8

  // ---- tile staging (TDM + manual im2col gather), double-buffered ----
  auto stage = [&](int kb, int buf) {
#if HAVE_TDM
    if (wave == 0) {
      if (!IM2COL) {
        tdm_load_tile((unsigned)(unsigned long long)(void*)&As[buf][0],
                      (unsigned long long)(const void*)(A + (size_t)m0 * K + kb),
                      K, M, 128, K);
      }
      tdm_load_tile((unsigned)(unsigned long long)(void*)&Bs[buf][0],
                    (unsigned long long)(const void*)(Bt + (size_t)n0 * K + kb),
                    K, N, 128, K);
    }
#else
    {
      const int n  = tid >> 1;
      const int c0 = (tid & 1) * 16;
#pragma unroll
      for (int e = 0; e < 16; ++e) {
        int gk2 = kb + c0 + e, gn = n0 + n;
        Bs[buf][(size_t)n * 32 + c0 + e] =
            (gk2 < K && gn < N) ? Bt[(size_t)gn * K + gk2] : (_Float16)0.f;
      }
      if (!IM2COL) {
#pragma unroll
        for (int i = 0; i < 16; ++i) {
          const int row = i * 8 + rbase;
          const int gm = m0 + row, gk = kb + kk;
          As[buf][(size_t)row * 32 + kk] =
              (gm < M && gk < K) ? A[(size_t)gm * K + gk] : (_Float16)0.f;
        }
      }
    }
#endif
    if (IM2COL) {
      // k-decompose ONCE per tile per thread (kk fixed)
      const int gk = kb + kk;
      const bool kok = gk < K;
      int ci = 0, kh = 0, kw = 0;
      if (kok) {
        ci = gk % Ci;
        int f = gk / Ci;
        kh = f / KW; kw = f % KW;
      }
#pragma unroll
      for (int i = 0; i < 16; ++i) {
        const int row = i * 8 + rbase;
        const int gm  = m0 + row;
        _Float16 val = (_Float16)0.f;
        if (kok && gm < M) {
          int img = gm >> hwo_sh;
          int hw  = gm & ((1 << hwo_sh) - 1);
          int oh  = hw >> wout_sh;
          int ow  = hw & ((1 << wout_sh) - 1);
          int ih  = oh * stride + kh - pad;
          int iw  = ow * stride + kw - pad;
          if ((unsigned)ih < (unsigned)Hin && (unsigned)iw < (unsigned)Win)
            val = A[(((size_t)img * Ci + ci) * Hin + ih) * Win + iw];
        }
        As[buf][(size_t)row * 32 + kk] = val;
      }
    }
  };

  v8f acc[2][4];
#pragma unroll
  for (int i = 0; i < 2; ++i)
#pragma unroll
    for (int j = 0; j < 4; ++j) acc[i][j] = (v8f){0.f,0.f,0.f,0.f,0.f,0.f,0.f,0.f};

  const int ntiles = (K + 31) >> 5;
  stage(0, 0);
  for (int t = 0; t < ntiles; ++t) {
    const int cur = t & 1;
#if HAVE_TDM
    __builtin_amdgcn_s_wait_tensorcnt((short)0);   // tile t DMA complete
#endif
    __syncthreads();                               // tile t visible, buf cur^1 free
    if (t + 1 < ntiles) stage((t + 1) << 5, cur ^ 1);   // issue-ahead
    v16h a0 = frag_a(As[cur], wm * 32);
    v16h a1 = frag_a(As[cur], wm * 32 + 16);
#pragma unroll
    for (int tj = 0; tj < 4; ++tj) {
      v16h b = frag_b(Bs[cur], wn * 64 + tj * 16);
      acc[0][tj] = __builtin_amdgcn_wmma_f32_16x16x32_f16(false, a0, false, b, (short)0, acc[0][tj], false, false);
      acc[1][tj] = __builtin_amdgcn_wmma_f32_16x16x32_f16(false, a1, false, b, (short)0, acc[1][tj], false, false);
    }
    __syncthreads();   // computing done before buf cur is re-staged next round
  }

  // -------- epilogue: C/D layout lane=N, VGPR r -> M = r + 8*half ----------
  const int lane = tid & 31;
  const int hlf  = lane >> 4;
  const int nl   = lane & 15;
#pragma unroll
  for (int ti = 0; ti < 2; ++ti)
#pragma unroll
    for (int tj = 0; tj < 4; ++tj)
#pragma unroll
      for (int r = 0; r < 8; ++r) {
        int row = m0 + wm * 32 + ti * 16 + r + 8 * hlf;
        int col = n0 + wn * 64 + tj * 16 + nl;
        if (row < M && col < N) {
          float v = acc[ti][tj][r];
          if (bias) v += bias[col];
          if (relu_out) v = fmaxf(v, 0.f);
          size_t o;
          if (IM2COL) {
            int img = row >> hwo_sh;
            int hw  = row & ((1 << hwo_sh) - 1);
            o = (((size_t)img * N + col) << hwo_sh) + hw;     // NCHW
          } else {
            o = (size_t)row * N + col;                        // row-major
          }
          if (Cf) Cf[o] = v;
          if (Ch) Ch[o] = (_Float16)v;
        }
      }
}

// ---------------------------------------------------------------------------
// Elementwise / reduction helpers
// ---------------------------------------------------------------------------
__global__ void k_f32_to_f16(const float* __restrict__ x, _Float16* __restrict__ y, size_t n) {
  for (size_t i = (size_t)blockIdx.x * 256 + threadIdx.x; i < n; i += (size_t)gridDim.x * 256)
    y[i] = (_Float16)x[i];
}

__global__ void k_relu_f16(const float* __restrict__ x, _Float16* __restrict__ y, size_t n) {
  for (size_t i = (size_t)blockIdx.x * 256 + threadIdx.x; i < n; i += (size_t)gridDim.x * 256)
    y[i] = (_Float16)fmaxf(x[i], 0.f);
}

__global__ void k_add_inplace(float* __restrict__ x, const float* __restrict__ h, size_t n) {
  for (size_t i = (size_t)blockIdx.x * 256 + threadIdx.x; i < n; i += (size_t)gridDim.x * 256)
    x[i] += h[i];
}

__global__ void k_relu_out(const float* __restrict__ x, float* __restrict__ y, size_t n) {
  for (size_t i = (size_t)blockIdx.x * 256 + threadIdx.x; i < n; i += (size_t)gridDim.x * 256)
    y[i] = fmaxf(x[i], 0.f);
}

// OIHW f32 -> transposed [Co][K] f16, k=(kh*KW+kw)*Ci+ci.  grid:(cdiv(K,256),Co)
__global__ void k_pack_conv_wT(const float* __restrict__ w, _Float16* __restrict__ out,
                               int Ci, int KH, int KW) {
  const int K = Ci * KH * KW;
  const int n = blockIdx.y;
  const int k = blockIdx.x * 256 + threadIdx.x;
  if (k >= K) return;
  int ci = k % Ci;
  int f  = k / Ci;
  int kh = f / KW, kw = f % KW;
  out[(size_t)n * K + k] = (_Float16)w[(((size_t)n * Ci + ci) * KH + kh) * KW + kw];
}

// [256,256] f32 row-major -> transposed [n][k] f16.  grid:(256),block:(256)
__global__ void k_transpose256_f16(const float* __restrict__ w, _Float16* __restrict__ out) {
  const int n = blockIdx.x, k = threadIdx.x;
  out[(size_t)n * 256 + k] = (_Float16)w[(size_t)k * 256 + n];
}

// training-mode BN stats over (N,H,W): one block per channel (HW = 1<<hw_sh)
__global__ __launch_bounds__(256) void k_bn_stats(const float* __restrict__ x,
    float* __restrict__ meanb, float* __restrict__ invb, int C, int NB, int hw_sh) {
  __shared__ float rs[256], rq[256];
  const int c = blockIdx.x, tid = threadIdx.x;
  const size_t cnt = (size_t)NB << hw_sh;
  float s = 0.f, q = 0.f;
  for (size_t i = tid; i < cnt; i += 256) {
    size_t n = i >> hw_sh, hw = i & (((size_t)1 << hw_sh) - 1);
    float v = x[(((n * C) + c) << hw_sh) + hw];
    s += v; q += v * v;
  }
  rs[tid] = s; rq[tid] = q; __syncthreads();
  for (int st = 128; st > 0; st >>= 1) {
    if (tid < st) { rs[tid] += rs[tid + st]; rq[tid] += rq[tid + st]; }
    __syncthreads();
  }
  if (tid == 0) {
    float m = rs[0] / (float)cnt;
    float v = rq[0] / (float)cnt - m * m;
    meanb[c] = m;
    invb[c]  = rsqrtf(v + 1e-5f);
  }
}

// y = gelu_exact(bn(x)) (+relu); writes f32 + f16 (NCHW, C=256)
__global__ void k_bn_gelu(const float* __restrict__ x, const float* __restrict__ meanb,
    const float* __restrict__ invb, const float* __restrict__ g, const float* __restrict__ be,
    float* __restrict__ yf, _Float16* __restrict__ yh, int hw_sh, size_t total, int relu_after) {
  for (size_t i = (size_t)blockIdx.x * 256 + threadIdx.x; i < total; i += (size_t)gridDim.x * 256) {
    int c = (int)((i >> hw_sh) & 255);
    float t = (x[i] - meanb[c]) * invb[c] * g[c] + be[c];
    t = 0.5f * t * (1.f + erff(t * 0.70710678118654752f));
    if (relu_after) t = fmaxf(t, 0.f);
    yf[i] = t;
    yh[i] = (_Float16)t;
  }
}

// ---------------------------------------------------------------------------
// Windowed attention (T=8, C=256, H=W=32; whw/wt powers of two)
// ---------------------------------------------------------------------------
DEVFN size_t tok_to_zidx(int tok, int c, int whw_sh, int wt_sh) {
  const int hn_sh = 5 - whw_sh, wn_sh = 5 - whw_sh, tn_sh = 3 - wt_sh;
  const int l_sh  = wt_sh + 2 * whw_sh;
  int win = tok >> l_sh, l = tok & ((1 << l_sh) - 1);
  int ww = l & ((1 << whw_sh) - 1);
  int wh = (l >> whw_sh) & ((1 << whw_sh) - 1);
  int wi = l >> (2 * whw_sh);
  int wn = win & ((1 << wn_sh) - 1); int t1 = win >> wn_sh;
  int hn = t1 & ((1 << hn_sh) - 1);  int t2 = t1 >> hn_sh;
  int tn = t2 & ((1 << tn_sh) - 1);  int b  = t2 >> tn_sh;
  int t = (tn << wt_sh) + wi;
  int h = (hn << whw_sh) + wh;
  int w = (wn << whw_sh) + ww;
  return ((((size_t)((b << 3) + t) << 8) + c) << 10) + (h << 5) + w;
}

__global__ void k_win_gather(const float* __restrict__ z, _Float16* __restrict__ xh,
    float* __restrict__ xf, int whw_sh, int wt_sh, size_t total) {
  for (size_t i = (size_t)blockIdx.x * 256 + threadIdx.x; i < total; i += (size_t)gridDim.x * 256) {
    int tok = (int)(i >> 8), c = (int)(i & 255);
    float v = z[tok_to_zidx(tok, c, whw_sh, wt_sh)];
    xh[i] = (_Float16)v;
    xf[i] = v;
  }
}

// one block per window; L = 1<<l_sh (8 or 16); 4 heads x 64
__global__ __launch_bounds__(256) void k_attn_core(const float* __restrict__ q,
    const float* __restrict__ k, const float* __restrict__ v,
    _Float16* __restrict__ o, int l_sh, float scale) {
  __shared__ float qs[16 * 256], ks[16 * 256], vs[16 * 256], sc[16 * 16];
  const int tid = threadIdx.x;
  const int L = 1 << l_sh;
  const size_t base = (size_t)blockIdx.x << (l_sh + 8);
  for (int i = tid; i < (L << 8); i += 256) {
    qs[i] = q[base + i]; ks[i] = k[base + i]; vs[i] = v[base + i];
  }
  __syncthreads();
  for (int h = 0; h < 4; ++h) {
    for (int p = tid; p < (L << l_sh); p += 256) {
      int l = p >> l_sh, m = p & (L - 1);
      float s = 0.f;
#pragma unroll
      for (int d = 0; d < 64; ++d)
        s += qs[(l << 8) + (h << 6) + d] * ks[(m << 8) + (h << 6) + d];
      sc[(l << 4) + m] = s * scale;
    }
    __syncthreads();
    if (tid < L) {
      float mx = -1e30f;
      for (int j = 0; j < L; ++j) mx = fmaxf(mx, sc[(tid << 4) + j]);
      float sum = 0.f;
      for (int j = 0; j < L; ++j) {
        float e = __expf(sc[(tid << 4) + j] - mx);
        sc[(tid << 4) + j] = e; sum += e;
      }
      float inv = 1.f / sum;
      for (int j = 0; j < L; ++j) sc[(tid << 4) + j] *= inv;
    }
    __syncthreads();
    for (int p = tid; p < (L << 6); p += 256) {
      int l = p >> 6, d = p & 63;
      float a = 0.f;
      for (int m = 0; m < L; ++m) a += sc[(l << 4) + m] * vs[(m << 8) + (h << 6) + d];
      o[base + (l << 8) + (h << 6) + d] = (_Float16)a;
    }
    __syncthreads();
  }
}

// LN(x + o) per token, scatter back to z (B,T,C,H,W)
__global__ __launch_bounds__(256) void k_add_ln_scatter(const float* __restrict__ xf,
    const float* __restrict__ of, const float* __restrict__ g, const float* __restrict__ b,
    float* __restrict__ z, int whw_sh, int wt_sh) {
  __shared__ float red[256];
  const int tok = blockIdx.x, c = threadIdx.x;
  float t = xf[((size_t)tok << 8) + c] + of[((size_t)tok << 8) + c];
  red[c] = t; __syncthreads();
  for (int s = 128; s > 0; s >>= 1) { if (c < s) red[c] += red[c + s]; __syncthreads(); }
  float m = red[0] * (1.f / 256.f); __syncthreads();
  float d = t - m;
  red[c] = d * d; __syncthreads();
  for (int s = 128; s > 0; s >>= 1) { if (c < s) red[c] += red[c + s]; __syncthreads(); }
  float var = red[0] * (1.f / 256.f);
  float y = d * rsqrtf(var + 1e-5f) * g[c] + b[c];
  z[tok_to_zidx(tok, c, whw_sh, wt_sh)] = y;
}

// ---------------------------------------------------------------------------
// Host orchestration
// ---------------------------------------------------------------------------
static inline int cdiv(int a, int b) { return (a + b - 1) / b; }
static inline int egrid(size_t n) {
  size_t g = (n + 255) / 256;
  return (int)(g > 4096 ? 4096 : g);
}

extern "C" void kernel_launch(void* const* d_in, const int* in_sizes, int n_in,
                              void* d_out, int out_size, void* d_ws, size_t ws_size,
                              hipStream_t stream) {
  (void)in_sizes; (void)n_in; (void)out_size; (void)ws_size;
  int cur = 0;
  const float* x_mo = (const float*)d_in[cur++];
  struct ConvP { const float *w1,*b1,*g1,*be1,*w2,*b2,*g2,*be2; } cp[2];
  for (int s = 0; s < 2; ++s) {
    cp[s].w1 = (const float*)d_in[cur++]; cp[s].b1 = (const float*)d_in[cur++];
    cp[s].g1 = (const float*)d_in[cur++]; cp[s].be1 = (const float*)d_in[cur++];
    cp[s].w2 = (const float*)d_in[cur++]; cp[s].b2 = (const float*)d_in[cur++];
    cp[s].g2 = (const float*)d_in[cur++]; cp[s].be2 = (const float*)d_in[cur++];
  }
  struct AttnP { const float *wq,*wk,*wv,*fc,*lng,*lnb; } ap[6];
  for (int i = 0; i < 6; ++i) {
    ap[i].wq = (const float*)d_in[cur++]; ap[i].wk = (const float*)d_in[cur++];
    ap[i].wv = (const float*)d_in[cur++]; ap[i].fc = (const float*)d_in[cur++];
    ap[i].lng = (const float*)d_in[cur++]; ap[i].lnb = (const float*)d_in[cur++];
  }
  struct ResP { const float *w3,*b3,*w1,*b1; } rp[2];
  for (int i = 0; i < 2; ++i) {
    rp[i].w3 = (const float*)d_in[cur++]; rp[i].b3 = (const float*)d_in[cur++];
    rp[i].w1 = (const float*)d_in[cur++]; rp[i].b1 = (const float*)d_in[cur++];
  }

  const int NB = 32, C = 256, NTOK = 32 * 32 * 32;

  char* ws = (char*)d_ws;
  size_t off = 0;
  auto carve = [&](size_t bytes) -> void* {
    off = (off + 255) & ~(size_t)255;
    void* p = ws + off;
    off += bytes;
    return p;
  };
  _Float16* xh    = (_Float16*)carve((size_t)NB * 3 * 128 * 128 * 2);
  float*    cbuf  = (float*)   carve((size_t)NB * C * 64 * 64 * 4);
  float*    zbuf  = (float*)   carve((size_t)NB * C * 64 * 64 * 4);
  _Float16* abuf  = (_Float16*)carve((size_t)NB * C * 64 * 64 * 2);
  _Float16* wbuf  = (_Float16*)carve((size_t)2 * 1024 * 1024 * 2);
  float*    meanb = (float*)   carve(256 * 4);
  float*    invb  = (float*)   carve(256 * 4);
  float*    qb    = (float*)   carve((size_t)NTOK * 256 * 4);
  float*    kbuf  = (float*)   carve((size_t)NTOK * 256 * 4);
  float*    vb    = (float*)   carve((size_t)NTOK * 256 * 4);
  float*    xf    = (float*)   carve((size_t)NTOK * 256 * 4);
  float*    of    = (float*)   carve((size_t)NTOK * 256 * 4);
  _Float16* ohb   = (_Float16*)carve((size_t)NTOK * 256 * 2);
  _Float16* h128  = (_Float16*)carve((size_t)NB * 128 * 1024 * 2);

  size_t nin = (size_t)NB * 3 * 128 * 128;
  k_f32_to_f16<<<egrid(nin), 256, 0, stream>>>(x_mo, xh, nin);

  // ---- downsampling conv stages ----
  const _Float16* actH = xh;
  int Ci = 3, Hin = 128, Win = 128;
  for (int s = 0; s < 2; ++s) {
    int Ho = Hin / 2, Wo = Win / 2;
    int wout_sh = (Wo == 64) ? 6 : 5;
    int hwo_sh  = 2 * wout_sh;
    int K1 = Ci * 25, M1 = NB * Ho * Wo;
    k_pack_conv_wT<<<dim3(cdiv(K1, 256), C), 256, 0, stream>>>(cp[s].w1, wbuf, Ci, 5, 5);
    k_gemm_wmma<1><<<dim3(cdiv(C, 128), cdiv(M1, 128)), 256, 0, stream>>>(
        actH, wbuf, cp[s].b1, cbuf, (_Float16*)nullptr,
        M1, C, K1, Ci, Hin, Win, wout_sh, hwo_sh, 5, 2, 2, 0);
    k_bn_stats<<<C, 256, 0, stream>>>(cbuf, meanb, invb, C, NB, hwo_sh);
    size_t tot1 = (size_t)NB * C << hwo_sh;
    k_bn_gelu<<<egrid(tot1), 256, 0, stream>>>(cbuf, meanb, invb, cp[s].g1, cp[s].be1,
                                               zbuf, abuf, hwo_sh, tot1, 0);
    int K2 = C * 9, M2 = M1;
    k_pack_conv_wT<<<dim3(cdiv(K2, 256), C), 256, 0, stream>>>(cp[s].w2, wbuf, C, 3, 3);
    k_gemm_wmma<1><<<dim3(cdiv(C, 128), cdiv(M2, 128)), 256, 0, stream>>>(
        abuf, wbuf, cp[s].b2, cbuf, (_Float16*)nullptr,
        M2, C, K2, C, Ho, Wo, wout_sh, hwo_sh, 3, 1, 1, 0);
    k_bn_stats<<<C, 256, 0, stream>>>(cbuf, meanb, invb, C, NB, hwo_sh);
    k_bn_gelu<<<egrid(tot1), 256, 0, stream>>>(cbuf, meanb, invb, cp[s].g2, cp[s].be2,
                                               zbuf, abuf, hwo_sh, tot1, 1);
    actH = abuf;
    Ci = C; Hin = Ho; Win = Wo;
  }
  // zbuf = z f32 [B,T,256,32,32]

  // ---- 6 windowed attention residual blocks ----
  const int whw_sh_cfg[3] = {2, 1, 0};   // whw = 4,2,1
  const int wt_sh_cfg[3]  = {0, 1, 3};   // wt  = 1,2,8
  size_t ntot = (size_t)NTOK * 256;
  for (int i = 0; i < 6; ++i) {
    int whw_sh = whw_sh_cfg[i % 3], wt_sh = wt_sh_cfg[i % 3];
    int l_sh = wt_sh + 2 * whw_sh;
    int nwin = NTOK >> l_sh;
    k_win_gather<<<egrid(ntot), 256, 0, stream>>>(zbuf, abuf, xf, whw_sh, wt_sh, ntot);
    k_transpose256_f16<<<256, 256, 0, stream>>>(ap[i].wq, wbuf);
    k_gemm_wmma<0><<<dim3(cdiv(256, 128), cdiv(NTOK, 128)), 256, 0, stream>>>(
        abuf, wbuf, (const float*)nullptr, qb, (_Float16*)nullptr,
        NTOK, 256, 256, 0, 0, 0, 0, 0, 1, 1, 0, 0);
    k_transpose256_f16<<<256, 256, 0, stream>>>(ap[i].wk, wbuf);
    k_gemm_wmma<0><<<dim3(cdiv(256, 128), cdiv(NTOK, 128)), 256, 0, stream>>>(
        abuf, wbuf, (const float*)nullptr, kbuf, (_Float16*)nullptr,
        NTOK, 256, 256, 0, 0, 0, 0, 0, 1, 1, 0, 0);
    k_transpose256_f16<<<256, 256, 0, stream>>>(ap[i].wv, wbuf);
    k_gemm_wmma<0><<<dim3(cdiv(256, 128), cdiv(NTOK, 128)), 256, 0, stream>>>(
        abuf, wbuf, (const float*)nullptr, vb, (_Float16*)nullptr,
        NTOK, 256, 256, 0, 0, 0, 0, 0, 1, 1, 0, 0);
    k_attn_core<<<nwin, 256, 0, stream>>>(qb, kbuf, vb, ohb, l_sh, 0.125f);
    k_transpose256_f16<<<256, 256, 0, stream>>>(ap[i].fc, wbuf);
    k_gemm_wmma<0><<<dim3(cdiv(256, 128), cdiv(NTOK, 128)), 256, 0, stream>>>(
        ohb, wbuf, (const float*)nullptr, of, (_Float16*)nullptr,
        NTOK, 256, 256, 0, 0, 0, 0, 0, 1, 1, 0, 0);
    k_add_ln_scatter<<<NTOK, 256, 0, stream>>>(xf, of, ap[i].lng, ap[i].lnb,
                                               zbuf, whw_sh, wt_sh);
  }

  // ---- residual stack ----
  size_t npix = (size_t)NB * C * 1024;
  for (int i = 0; i < 2; ++i) {
    k_relu_f16<<<egrid(npix), 256, 0, stream>>>(zbuf, abuf, npix);
    int K3 = C * 9, M3 = NB * 1024;
    k_pack_conv_wT<<<dim3(cdiv(K3, 256), 128), 256, 0, stream>>>(rp[i].w3, wbuf, C, 3, 3);
    k_gemm_wmma<1><<<dim3(cdiv(128, 128), cdiv(M3, 128)), 256, 0, stream>>>(
        abuf, wbuf, rp[i].b3, cbuf, h128,
        M3, 128, K3, C, 32, 32, 5, 10, 3, 1, 1, 1);
    k_pack_conv_wT<<<dim3(cdiv(128, 256), C), 256, 0, stream>>>(rp[i].w1, wbuf, 128, 1, 1);
    k_gemm_wmma<1><<<dim3(cdiv(C, 128), cdiv(M3, 128)), 256, 0, stream>>>(
        h128, wbuf, rp[i].b1, of, (_Float16*)nullptr,
        M3, C, 128, 128, 32, 32, 5, 10, 1, 1, 0, 0);
    k_add_inplace<<<egrid(npix), 256, 0, stream>>>(zbuf, of, npix);
  }
  k_relu_out<<<egrid(npix), 256, 0, stream>>>(zbuf, (float*)d_out, npix);
}
